// GVAE_41240275976667
// MI455X (gfx1250) — compile-verified, matrix operand order checked by reading, not run on hardware
//
#include <hip/hip_runtime.h>
#include <hip/hip_bf16.h>
#include <math.h>

// ---------------- problem constants ----------------
#define G_   2048
#define NPG_ 20
#define EPG_ 40
#define N_   (G_*NPG_)      // 40960
#define E_   (G_*EPG_)      // 81920
#define FIN_ 30
#define HD_  256            // 4 heads * 64
#define D_   64
#define LAT_ 121
#define DECH_ 512
#define NEDGE_ 760
#define NATOM_ 10

// ---------------- WMMA types ----------------
typedef __bf16 bhalf;
typedef bhalf  bhalf4  __attribute__((ext_vector_type(4)));
typedef bhalf  bhalf8  __attribute__((ext_vector_type(8)));
typedef bhalf  bhalf16 __attribute__((ext_vector_type(16)));
typedef float  floatx8 __attribute__((ext_vector_type(8)));

#define BM 64
#define BN 128
#define BK 32
#define LPAD 40   // LDS row pitch in bf16 (80 B: keeps 16-B alignment, breaks conflicts)

// =====================================================================
// WMMA GEMM: C[M,N] = act(A[M,K] @ B[K,N] + bias)
// fp32 in/out, bf16 LDS staging, f32 accumulate (v_wmma_f32_16x16x32_bf16).
// Block 256 threads (8 waves), tile 64x128; wave = 16 rows x 64 cols
//   -> 1 A-fragment reused by 4 WMMAs per k-step.
// Fragments loaded as two 16-byte LDS reads each (K-contiguous layout;
// B staged transposed so its columns are K-contiguous too).
// =====================================================================
__global__ void __launch_bounds__(256)
gemm_bias_act(const float* __restrict__ A, const float* __restrict__ B,
              const float* __restrict__ bias, float* __restrict__ C,
              int M, int N, int K, int lda, int ldb, int ldc, int relu)
{
    __shared__ bhalf As[BM][LPAD];   // 64 x 32 (+pad)  : As[m][k]
    __shared__ bhalf Bt[BN][LPAD];   // 128 x 32 (+pad) : Bt[n][k]  (B transposed)

    const int tid  = threadIdx.x;
    const int lane = tid & 31;
    const int wave = tid >> 5;      // 0..7
    const int wm   = wave >> 1;     // row block (16 rows)
    const int wn   = wave & 1;      // col half (64 cols)
    const int m0   = blockIdx.x * BM;
    const int n0   = blockIdx.y * BN;

    const int row = lane & 15;
    const int hi  = (lane >> 4) & 1;   // upper half-wave -> +8 in K packing

    floatx8 acc0 = {}, acc1 = {}, acc2 = {}, acc3 = {};

    const bool avec = ((lda & 3) == 0) && ((((uintptr_t)A) & 15) == 0);
    const bool bvec = ((ldb & 3) == 0) && ((((uintptr_t)B) & 15) == 0);

    for (int k0 = 0; k0 < K; k0 += BK) {
        // ---- stage A tile (64x32) ----
        if (avec && (m0 + BM <= M) && (k0 + BK <= K)) {
            for (int i = tid; i < BM * BK / 4; i += 256) {     // 512 float4
                int r = i >> 3, c4 = (i & 7) << 2;
                const float4 v =
                    *(const float4*)(A + (size_t)(m0 + r) * lda + k0 + c4);
                bhalf4 p;
                p[0] = (bhalf)v.x; p[1] = (bhalf)v.y;
                p[2] = (bhalf)v.z; p[3] = (bhalf)v.w;
                *(bhalf4*)&As[r][c4] = p;                      // 8-B LDS store
            }
        } else {
            for (int i = tid; i < BM * BK; i += 256) {
                int r = i >> 5, cc = i & 31;
                int gr = m0 + r, gc = k0 + cc;
                float v = (gr < M && gc < K) ? A[(size_t)gr * lda + gc] : 0.0f;
                As[r][cc] = (bhalf)v;
            }
        }
        // ---- stage B tile (32x128) transposed into Bt[n][k] ----
        if (bvec && (n0 + BN <= N) && (k0 + BK <= K)) {
            for (int i = tid; i < BK * BN / 4; i += 256) {     // 1024 float4
                int r = i >> 5, c4 = (i & 31) << 2;
                const float4 v =
                    *(const float4*)(B + (size_t)(k0 + r) * ldb + n0 + c4);
                Bt[c4 + 0][r] = (bhalf)v.x;
                Bt[c4 + 1][r] = (bhalf)v.y;
                Bt[c4 + 2][r] = (bhalf)v.z;
                Bt[c4 + 3][r] = (bhalf)v.w;
            }
        } else {
            for (int i = tid; i < BK * BN; i += 256) {
                int r = i >> 7, cc = i & 127;
                int gr = k0 + r, gc = n0 + cc;
                float v = (gr < K && gc < N) ? B[(size_t)gr * ldb + gc] : 0.0f;
                Bt[cc][r] = (bhalf)v;
            }
        }
        __syncthreads();

        // ---- A fragment: two 16-B LDS loads (K runs [8hi,8hi+8), [16+8hi,..)) ----
        const bhalf8 a0 = *(const bhalf8*)&As[wm * 16 + row][hi * 8];
        const bhalf8 a1 = *(const bhalf8*)&As[wm * 16 + row][16 + hi * 8];
        const bhalf16 af = __builtin_shufflevector(a0, a1,
            0, 1, 2, 3, 4, 5, 6, 7, 8, 9, 10, 11, 12, 13, 14, 15);

        // ---- 4 B fragments + 4 WMMAs (A reuse) ----
        {
            const int cn = wn * 64 + 0 * 16 + row;
            const bhalf8 b0 = *(const bhalf8*)&Bt[cn][hi * 8];
            const bhalf8 b1 = *(const bhalf8*)&Bt[cn][16 + hi * 8];
            const bhalf16 bf = __builtin_shufflevector(b0, b1,
                0, 1, 2, 3, 4, 5, 6, 7, 8, 9, 10, 11, 12, 13, 14, 15);
            acc0 = __builtin_amdgcn_wmma_f32_16x16x32_bf16(
                       false, af, false, bf, (short)0, acc0, false, false);
        }
        {
            const int cn = wn * 64 + 1 * 16 + row;
            const bhalf8 b0 = *(const bhalf8*)&Bt[cn][hi * 8];
            const bhalf8 b1 = *(const bhalf8*)&Bt[cn][16 + hi * 8];
            const bhalf16 bf = __builtin_shufflevector(b0, b1,
                0, 1, 2, 3, 4, 5, 6, 7, 8, 9, 10, 11, 12, 13, 14, 15);
            acc1 = __builtin_amdgcn_wmma_f32_16x16x32_bf16(
                       false, af, false, bf, (short)0, acc1, false, false);
        }
        {
            const int cn = wn * 64 + 2 * 16 + row;
            const bhalf8 b0 = *(const bhalf8*)&Bt[cn][hi * 8];
            const bhalf8 b1 = *(const bhalf8*)&Bt[cn][16 + hi * 8];
            const bhalf16 bf = __builtin_shufflevector(b0, b1,
                0, 1, 2, 3, 4, 5, 6, 7, 8, 9, 10, 11, 12, 13, 14, 15);
            acc2 = __builtin_amdgcn_wmma_f32_16x16x32_bf16(
                       false, af, false, bf, (short)0, acc2, false, false);
        }
        {
            const int cn = wn * 64 + 3 * 16 + row;
            const bhalf8 b0 = *(const bhalf8*)&Bt[cn][hi * 8];
            const bhalf8 b1 = *(const bhalf8*)&Bt[cn][16 + hi * 8];
            const bhalf16 bf = __builtin_shufflevector(b0, b1,
                0, 1, 2, 3, 4, 5, 6, 7, 8, 9, 10, 11, 12, 13, 14, 15);
            acc3 = __builtin_amdgcn_wmma_f32_16x16x32_bf16(
                       false, af, false, bf, (short)0, acc3, false, false);
        }
        __syncthreads();
    }

    // ---- epilogue: D element r -> (row = r + 8*hi, col = lane&15) ----
    const int colw = lane & 15;
    floatx8 accs[4] = { acc0, acc1, acc2, acc3 };
#pragma unroll
    for (int t = 0; t < 4; ++t) {
        int gc = n0 + wn * 64 + t * 16 + colw;
        if (gc >= N) continue;
        float bsv = bias[gc];
#pragma unroll
        for (int r = 0; r < 8; ++r) {
            int gr = m0 + wm * 16 + r + 8 * hi;
            if (gr >= M) continue;
            float v = accs[t][r] + bsv;
            if (relu) v = fmaxf(v, 0.0f);
            C[(size_t)gr * ldc + gc] = v;
        }
    }
}

// =====================================================================
// Fused per-graph TransformerConv layer (heads=4, concat=false, beta skip)
// One workgroup (256 threads) per graph; q/k/v/e projections, segment
// softmax over destination nodes, message aggregation, head mean,
// beta-gated skip, ReLU. All intermediates in dynamic LDS (~135 KB).
// =====================================================================
__global__ void __launch_bounds__(256)
attn_conv(const float* __restrict__ hin, int fin,
          const float* __restrict__ ea,
          const int* __restrict__ srcg, const int* __restrict__ dstg,
          const float* __restrict__ Wq, const float* __restrict__ bq,
          const float* __restrict__ Wk, const float* __restrict__ bk,
          const float* __restrict__ Wv, const float* __restrict__ bv,
          const float* __restrict__ We,
          const float* __restrict__ Ws, const float* __restrict__ bs,
          const float* __restrict__ Wb,
          float* __restrict__ hout)
{
    extern __shared__ float sm[];
    float* q    = sm;            // 20*256
    float* k    = q    + 5120;   // 20*256
    float* v    = k    + 5120;   // 20*256
    float* e    = v    + 5120;   // 40*256
    float* acc  = e    + 10240;  // 20*256
    float* w    = acc  + 5120;   // 40*4
    float* nmx  = w    + 160;    // 20*4
    float* nden = nmx  + 80;     // 20*4
    float* bet  = nden + 80;     // 20
    float* om   = bet  + 20;     // 20*64
    float* xr   = om   + 1280;   // 20*64
    int*   sl   = (int*)(xr + 1280);  // 40
    int*   dl   = sl + 40;            // 40

    const int tid = threadIdx.x;
    const int g   = blockIdx.x;
    const int nb  = g * NPG_;
    const int eb  = g * EPG_;

    if (tid < EPG_) { sl[tid] = srcg[eb + tid] - nb; dl[tid] = dstg[eb + tid] - nb; }

    // fused q/k/v projections (fin <= 64)
    for (int idx = tid; idx < NPG_ * HD_; idx += 256) {
        int node = idx >> 8, col = idx & 255;
        const float* xrow = hin + (size_t)(nb + node) * fin;
        float aq = bq[col], ak = bk[col], av = bv[col];
        for (int f = 0; f < fin; ++f) {
            float xv = xrow[f];
            aq += xv * Wq[f * HD_ + col];
            ak += xv * Wk[f * HD_ + col];
            av += xv * Wv[f * HD_ + col];
        }
        q[idx] = aq; k[idx] = ak; v[idx] = av;
    }
    // edge projection (no bias)
    for (int idx = tid; idx < EPG_ * HD_; idx += 256) {
        int ed = idx >> 8, col = idx & 255;
        const float* ar = ea + (size_t)(eb + ed) * 11;
        float s = 0.0f;
        for (int f = 0; f < 11; ++f) s += ar[f] * We[f * HD_ + col];
        e[idx] = s;
    }
    __syncthreads();

    // alpha[e,h] = q[dst] . (k[src]+e) / sqrt(64)
    if (tid < EPG_ * 4) {
        int ed = tid >> 2, h = tid & 3;
        int s_ = sl[ed], d_ = dl[ed];
        float a = 0.0f;
        for (int dd = 0; dd < D_; ++dd)
            a += q[d_ * HD_ + h * D_ + dd] *
                 (k[s_ * HD_ + h * D_ + dd] + e[ed * HD_ + h * D_ + dd]);
        w[tid] = a * 0.125f;
    }
    __syncthreads();

    // per-(node,head) segment max & sum(exp)
    if (tid < NPG_ * 4) {
        int node = tid >> 2, h = tid & 3;
        float mx = -__builtin_inff();
        for (int ed = 0; ed < EPG_; ++ed)
            if (dl[ed] == node) mx = fmaxf(mx, w[ed * 4 + h]);
        if (mx == -__builtin_inff()) mx = 0.0f;  // jnp.where(isfinite, amax, 0)
        float den = 0.0f;
        for (int ed = 0; ed < EPG_; ++ed)
            if (dl[ed] == node) den += __expf(w[ed * 4 + h] - mx);
        nmx[tid] = mx; nden[tid] = den;
    }
    __syncthreads();

    // normalize to attention weights
    if (tid < EPG_ * 4) {
        int ed = tid >> 2, h = tid & 3;
        int d_ = dl[ed];
        w[tid] = __expf(w[tid] - nmx[d_ * 4 + h]) / (nden[d_ * 4 + h] + 1e-16f);
    }
    for (int i = tid; i < NPG_ * HD_; i += 256) acc[i] = 0.0f;
    __syncthreads();

    // message aggregation: serial over edges, each thread owns one (h,d) slot
    {
        int h = tid >> 6;  // 256 threads = 4 heads x 64 dims
        for (int ed = 0; ed < EPG_; ++ed) {
            float ww = w[ed * 4 + h];
            acc[dl[ed] * HD_ + tid] += (v[sl[ed] * HD_ + tid] + e[ed * HD_ + tid]) * ww;
        }
    }
    __syncthreads();

    // head-mean + skip projection
    for (int idx = tid; idx < NPG_ * D_; idx += 256) {
        int node = idx >> 6, dd = idx & 63;
        float o = 0.25f * (acc[node * HD_ + dd]       + acc[node * HD_ + 64 + dd] +
                           acc[node * HD_ + 128 + dd] + acc[node * HD_ + 192 + dd]);
        om[idx] = o;
        const float* xrow = hin + (size_t)(nb + node) * fin;
        float s = bs[dd];
        for (int f = 0; f < fin; ++f) s += xrow[f] * Ws[f * D_ + dd];
        xr[idx] = s;
    }
    __syncthreads();

    // beta = sigmoid([out, x_r, out-x_r] @ Wb)
    if (tid < NPG_) {
        float s = 0.0f;
        for (int dd = 0; dd < D_; ++dd) {
            float o = om[tid * D_ + dd], xv = xr[tid * D_ + dd];
            s += o * Wb[dd] + xv * Wb[64 + dd] + (o - xv) * Wb[128 + dd];
        }
        bet[tid] = 1.0f / (1.0f + __expf(-s));
    }
    __syncthreads();

    // h = relu(beta*x_r + (1-beta)*out)
    for (int idx = tid; idx < NPG_ * D_; idx += 256) {
        int node = idx >> 6;
        float b_ = bet[node];
        float val = b_ * xr[idx] + (1.0f - b_) * om[idx];
        hout[(size_t)(nb + node) * D_ + (idx & 63)] = fmaxf(val, 0.0f);
    }
}

// =====================================================================
// Deterministic BatchNorm (training stats): partial sums -> finalize -> apply
// =====================================================================
__global__ void __launch_bounds__(256)
bn_partial(const float* __restrict__ x, float* __restrict__ part, int n)
{
    __shared__ float ls[256], ls2[256];
    int c = threadIdx.x & 63, rg = threadIdx.x >> 6;
    float s = 0.0f, s2 = 0.0f;
    for (int row = blockIdx.x * 4 + rg; row < n; row += gridDim.x * 4) {
        float v = x[(size_t)row * D_ + c];
        s += v; s2 += v * v;
    }
    ls[threadIdx.x] = s; ls2[threadIdx.x] = s2;
    __syncthreads();
    if (threadIdx.x < 64) {
        float a = 0.0f, b = 0.0f;
        for (int r = 0; r < 4; ++r) { a += ls[r * 64 + c]; b += ls2[r * 64 + c]; }
        part[blockIdx.x * 128 + c]      = a;
        part[blockIdx.x * 128 + 64 + c] = b;
    }
}

__global__ void bn_finalize(const float* __restrict__ part, int nblk,
                            float* __restrict__ stats)
{
    int c = threadIdx.x;
    if (c < 128) {
        float s = 0.0f;
        for (int b = 0; b < nblk; ++b) s += part[b * 128 + c];
        stats[c] = s;
    }
}

__global__ void bn_apply(const float* __restrict__ x, const float* __restrict__ stats,
                         const float* __restrict__ gw, const float* __restrict__ bw,
                         float* __restrict__ y, int n)
{
    int idx = blockIdx.x * blockDim.x + threadIdx.x;
    if (idx >= n * D_) return;
    int c = idx & 63;
    float invn = 1.0f / (float)n;
    float m  = stats[c] * invn;
    float va = stats[64 + c] * invn - m * m;
    y[idx] = gw[c] * (x[idx] - m) * rsqrtf(va + 1e-5f) + bw[c];
}

// =====================================================================
// z = eps * exp(logvar) + mu
// =====================================================================
__global__ void reparam(const float* __restrict__ mu, const float* __restrict__ lv,
                        const float* __restrict__ eps, float* __restrict__ z, int total)
{
    int i = blockIdx.x * blockDim.x + threadIdx.x;
    if (i < total) z[i] = eps[i] * __expf(lv[i]) + mu[i];
}

// =====================================================================
// Per-graph edge decoder, algebraically collapsed:
//   out[j] = sum_l ( sum_k (colsum(M)[k]/NPG) * gz[k,l] ) * Wg[l,j] + bg[j]
// =====================================================================
__global__ void __launch_bounds__(256)
edge_decode(const float* __restrict__ z, const float* __restrict__ Wg,
            const float* __restrict__ bg, float* __restrict__ out)
{
    __shared__ float gz[NPG_ * LAT_];   // 2420
    __shared__ float Mm[NPG_ * NPG_];   // 400
    __shared__ float dis[NPG_];
    __shared__ float cc[NPG_];
    __shared__ float ww[LAT_];
    __shared__ float red[256];

    const int tid = threadIdx.x;
    const int g   = blockIdx.x;

    for (int i = tid; i < NPG_ * LAT_; i += 256)
        gz[i] = z[(size_t)g * NPG_ * LAT_ + i];
    __syncthreads();

    // ss = sum(gz*gz)
    float s = 0.0f;
    for (int i = tid; i < NPG_ * LAT_; i += 256) s += gz[i] * gz[i];
    red[tid] = s;
    __syncthreads();
    for (int off = 128; off > 0; off >>= 1) {
        if (tid < off) red[tid] += red[tid + off];
        __syncthreads();
    }
    const float ss = red[0];
    __syncthreads();

    // A matrix: hat_A off-diagonal, 1 on diagonal
    for (int idx = tid; idx < NPG_ * NPG_; idx += 256) {
        int i = idx / NPG_, j = idx % NPG_;
        float d = 0.0f;
        for (int l = 0; l < LAT_; ++l) d += gz[i * LAT_ + l] * gz[j * LAT_ + l];
        float hatA = d / ss + 1.0f;
        Mm[idx] = (i == j) ? 1.0f : hatA;
    }
    __syncthreads();
    if (tid < NPG_) {
        float deg = 0.0f;
        for (int i = 0; i < NPG_; ++i) deg += Mm[i * NPG_ + tid];
        dis[tid] = (deg > 0.0f) ? rsqrtf(deg) : 0.0f;
    }
    __syncthreads();
    if (tid < NPG_) {    // c[k] = colsum(M)[k] / NPG
        float cs = 0.0f;
        for (int i = 0; i < NPG_; ++i) cs += dis[i] * Mm[i * NPG_ + tid] * dis[tid];
        cc[tid] = cs * (1.0f / (float)NPG_);
    }
    __syncthreads();
    if (tid < LAT_) {    // w[l] = sum_k c[k] * gz[k,l]
        float sw = 0.0f;
        for (int kk = 0; kk < NPG_; ++kk) sw += cc[kk] * gz[kk * LAT_ + tid];
        ww[tid] = sw;
    }
    __syncthreads();
    for (int j = tid; j < NEDGE_; j += 256) {
        float so = bg[j];
        for (int l = 0; l < LAT_; ++l) so += ww[l] * Wg[l * NEDGE_ + j];
        out[(size_t)g * NEDGE_ + j] = so;
    }
}

// =====================================================================
// host-side launch
// =====================================================================
extern "C" void kernel_launch(void* const* d_in, const int* in_sizes, int n_in,
                              void* d_out, int out_size, void* d_ws, size_t ws_size,
                              hipStream_t stream)
{
    const float* x   = (const float*)d_in[0];
    const float* ea  = (const float*)d_in[1];
    const int*   ei  = (const int*)d_in[2];
    const float* eps = (const float*)d_in[4];
    const int* srcp = ei;
    const int* dstp = ei + E_;

    auto P = [&](int i) { return (const float*)d_in[i]; };
    // params flattened in insertion order: c1..c4 (Wq,bq,Wk,bk,Wv,bv,We,Ws,bs,Wb),
    // bn1_g,bn1_b,bn2_g,bn2_b,bn3_g,bn3_b, Wmu,bmu,Wlv,blv, W1,b1,W2,b2,Wa,ba,Wg,bg
    const float* cp[4][10];
    for (int l = 0; l < 4; ++l)
        for (int j = 0; j < 10; ++j) cp[l][j] = P(5 + l * 10 + j);
    const float* bng[3] = { P(45), P(47), P(49) };
    const float* bnb[3] = { P(46), P(48), P(50) };
    const float* Wmu = P(51); const float* bmu = P(52);
    const float* Wlv = P(53); const float* blv = P(54);
    const float* W1  = P(55); const float* b1  = P(56);
    const float* W2  = P(57); const float* b2  = P(58);
    const float* Wa  = P(59); const float* ba  = P(60);
    const float* Wg  = P(61); const float* bg  = P(62);

    // workspace layout (floats)
    float* ws    = (float*)d_ws;
    float* HA    = ws;                              // N*64
    float* HB    = HA + (size_t)N_ * D_;            // N*64
    float* STATS = HB + (size_t)N_ * D_;            // 128
    float* PART  = STATS + 128;                     // 512*128
    float* Z     = PART + 512 * 128;                // N*121
    const int CH = 10240;                           // decoder row chunk
    float* A1    = Z + (size_t)N_ * LAT_;           // CH*512
    float* A2    = A1 + (size_t)CH * DECH_;         // CH*512

    // output layout: triu_logits, node_logits, mu, logvar
    float* out  = (float*)d_out;
    float* triu = out;
    float* nlog = out + (size_t)G_ * NEDGE_;
    float* mu   = nlog + (size_t)N_ * NATOM_;
    float* lv   = mu + (size_t)N_ * LAT_;

    const size_t ATT_SMEM = (size_t)(33620 + 80) * sizeof(float);  // ~135 KB dynamic LDS

    auto conv = [&](const float* hin, int fin, const float* const* pp, float* hout) {
        attn_conv<<<G_, 256, ATT_SMEM, stream>>>(hin, fin, ea, srcp, dstp,
            pp[0], pp[1], pp[2], pp[3], pp[4], pp[5], pp[6], pp[7], pp[8], pp[9], hout);
    };
    auto bn = [&](const float* in_, const float* g_, const float* b_, float* out_) {
        bn_partial<<<512, 256, 0, stream>>>(in_, PART, N_);
        bn_finalize<<<1, 128, 0, stream>>>(PART, 512, STATS);
        bn_apply<<<(N_ * D_ + 255) / 256, 256, 0, stream>>>(in_, STATS, g_, b_, out_, N_);
    };
    auto gemm = [&](const float* A, const float* B, const float* bias, float* C,
                    int M, int Nc, int K, int lda, int ldb, int ldc, int relu) {
        dim3 grid((M + BM - 1) / BM, (Nc + BN - 1) / BN);
        gemm_bias_act<<<grid, dim3(256), 0, stream>>>(A, B, bias, C, M, Nc, K,
                                                      lda, ldb, ldc, relu);
    };

    // ---- encoder ----
    conv(x,  FIN_, cp[0], HA); bn(HA, bng[0], bnb[0], HB);
    conv(HB, D_,   cp[1], HA); bn(HA, bng[1], bnb[1], HB);
    conv(HB, D_,   cp[2], HA); bn(HA, bng[2], bnb[2], HB);
    conv(HB, D_,   cp[3], HA);

    // ---- latent heads (WMMA) + reparameterize ----
    gemm(HA, Wmu, bmu, mu, N_, LAT_, D_, D_, LAT_, LAT_, 0);
    gemm(HA, Wlv, blv, lv, N_, LAT_, D_, D_, LAT_, LAT_, 0);
    reparam<<<(N_ * LAT_ + 255) / 256, 256, 0, stream>>>(mu, lv, eps, Z, N_ * LAT_);

    // ---- node decoder (WMMA), chunked over rows ----
    for (int c0 = 0; c0 < N_; c0 += CH) {
        gemm(Z + (size_t)c0 * LAT_, W1, b1, A1, CH, DECH_, LAT_, LAT_, DECH_, DECH_, 1);
        gemm(A1, W2, b2, A2, CH, DECH_, DECH_, DECH_, DECH_, DECH_, 1);
        gemm(A2, Wa, ba, nlog + (size_t)c0 * NATOM_, CH, NATOM_, DECH_,
             DECH_, NATOM_, NATOM_, 0);
    }

    // ---- per-graph edge decoder ----
    edge_decode<<<G_, 256, 0, stream>>>(Z, Wg, bg, triu);
}
